// APPNPDecoder_32959579030046
// MI455X (gfx1250) — matrix operand order repaired and checked by reading, not compiled, not run on hardware
//
#include <hip/hip_runtime.h>
#include <stdint.h>

// ---------------------------------------------------------------------------
// APPNP decoder for MI455X (gfx1250, wave32).
//   GEMMs  : v_wmma_f32_16x16x32_bf16 (bf16 in, f32 accumulate).
//            A and B^T staged in padded LDS so every fragment is exactly
//            two 16B ds_load_b128 per matrix, bank-conflict free.
//   Prop   : wave-per-edge gather, global_atomic_add_f32 scatter (L2-resident).
// ---------------------------------------------------------------------------

#define N_NODES 50000
#define N_EDGES 800000
#define IN_CH   64
#define HID_CH  128
#define OUT_CH  64
#define K_STEPS 10
#define ALPHA   0.1f

typedef __attribute__((ext_vector_type(16))) __bf16 v16bf;
typedef __attribute__((ext_vector_type(8)))  __bf16 v8bf;
typedef __attribute__((ext_vector_type(8)))  float  v8f;

// ------------------------------- degree / norm -----------------------------

__global__ void deg_init_kernel(float* __restrict__ dinv, int n) {
    int i = blockIdx.x * blockDim.x + threadIdx.x;
    if (i < n) dinv[i] = 1.0f;             // self loop contributes 1
}

__global__ void deg_accum_kernel(const long long* __restrict__ dst,
                                 float* __restrict__ dinv, int nE) {
    int e = blockIdx.x * blockDim.x + threadIdx.x;
    if (e < nE) atomicAdd(&dinv[(int)dst[e]], 1.0f);
}

__global__ void deg_fin_kernel(float* __restrict__ dinv, int n) {
    int i = blockIdx.x * blockDim.x + threadIdx.x;
    if (i < n) dinv[i] = rsqrtf(fmaxf(dinv[i], 1.0f));
}

__global__ void norm_kernel(const long long* __restrict__ src,
                            const long long* __restrict__ dst,
                            const float* __restrict__ dinv,
                            float* __restrict__ norm, int nE, int nN) {
    int e = blockIdx.x * blockDim.x + threadIdx.x;
    if (e >= nE + nN) return;
    int s, d;
    if (e < nE) { s = (int)src[e]; d = (int)dst[e]; }
    else        { s = d = e - nE; }
    norm[e] = dinv[s] * dinv[d];
}

// ------------------------------- WMMA GEMM ---------------------------------
// C[M x NC] = A[M x K] * B[K x NC] + bias. One 16x16 tile per wave,
// blockDim.x = 32 * (NC/16), K stepped by 32 through v_wmma_f32_16x16x32_bf16.
// RELUA applies relu to A while converting f32 -> bf16 into LDS.
//
// LDS layout: rows padded by 8 bf16 (16 B) so the per-lane row stride is
// 16 B-shifted across the 64 LDS banks -> conflict-free ds_load_b128.

template <int K, int NC, bool RELUA>
__global__ void gemm_wmma_kernel(const float* __restrict__ A,
                                 const float* __restrict__ B,
                                 const float* __restrict__ bias,
                                 float* __restrict__ C) {
    constexpr int PAD = 8;                 // bf16 elements (16 bytes)
    constexpr int SA  = K + PAD;           // sA row stride
    constexpr int SB  = K + PAD;           // sBt row stride
    __shared__ __align__(16) __bf16 sA [16 * SA];
    __shared__ __align__(16) __bf16 sBt[NC * SB];

    const int tid  = threadIdx.x;
    const int nthr = blockDim.x;
    const int row0 = blockIdx.x * 16;

    // Stage A tile (16 x K), f32 -> bf16, optional ReLU.
    for (int i = tid; i < 16 * K; i += nthr) {
        int r = i / K, k = i - r * K;
        float v = A[(size_t)(row0 + r) * K + k];
        if (RELUA) v = v > 0.0f ? v : 0.0f;
        sA[r * SA + k] = (__bf16)v;
    }
    // Stage B transposed: sBt[n][k] (coalesced global read, hot in L2).
    for (int i = tid; i < K * NC; i += nthr) {
        int k = i / NC, n = i - k * NC;
        sBt[n * SB + k] = (__bf16)B[i];
    }
    __syncthreads();

    const int wave    = tid >> 5;
    const int lane    = tid & 31;
    const int colBase = wave * 16;
    const int half    = lane >> 4;   // 0: lanes 0-15, 1: lanes 16-31
    const int l16     = lane & 15;

    const __bf16* aRow = &sA [l16 * SA];              // row M = l16
    const __bf16* bRow = &sBt[(colBase + l16) * SB];  // col N = colBase+l16

    v8f acc = {};
#pragma unroll
    for (int kc = 0; kc < K; kc += 32) {
        // A fragment (16-bit 16x32 ISA layout): lanes 0-15 hold K=kc+0..7 and
        // kc+16..23; lanes 16-31 hold K=kc+8..15 and kc+24..31.
        const int kbA = kc + half * 8;
        union { v16bf v; v8bf h[2]; } au;
        au.h[0] = *(const v8bf*)(aRow + kbA);        // 16B ds_load_b128
        au.h[1] = *(const v8bf*)(aRow + kbA + 16);   // 16B ds_load_b128
        // B fragment: same K order as A (lanes 0-15 K=kc+0..7,16..23; lanes 16-31 K=kc+8..15,24..31).
        const int kbB = kc + half * 8;
        union { v16bf v; v8bf h[2]; } bu;
        bu.h[0] = *(const v8bf*)(bRow + kbB);        // 16B ds_load_b128
        bu.h[1] = *(const v8bf*)(bRow + kbB + 16);   // 16B ds_load_b128
        acc = __builtin_amdgcn_wmma_f32_16x16x32_bf16(
            false, au.v, false, bu.v, (short)0, acc, false, false);
    }

    // C/D layout: lane column = l16; VGPR r holds row half*8 + r.
    const int col = colBase + l16;
    const float bz = bias[col];
#pragma unroll
    for (int r = 0; r < 8; ++r) {
        int row = row0 + half * 8 + r;
        C[(size_t)row * NC + col] = acc[r] + bz;
    }
}

// ------------------------------ propagation --------------------------------
// One wave per (virtual) edge; 32 lanes cover CH channels with CH/32 floats
// each. Gather z[src]*norm, scatter-add into agg[dst] with f32 atomics.

template <int CH>
__global__ void prop_scatter_kernel(const float* __restrict__ z,
                                    const long long* __restrict__ src,
                                    const long long* __restrict__ dst,
                                    const float* __restrict__ norm,
                                    float* __restrict__ agg,
                                    int nE, int nN) {
    constexpr int VEC = CH / 32;
    int gid  = blockIdx.x * blockDim.x + threadIdx.x;
    int e    = gid >> 5;
    int lane = gid & 31;
    if (e >= nE + nN) return;

    int s, d;
    if (e < nE) { s = (int)src[e]; d = (int)dst[e]; }
    else        { s = d = e - nE; }

    const float w = norm[e];
    const float* zp = z   + (size_t)s * CH + lane * VEC;
    float*       ap = agg + (size_t)d * CH + lane * VEC;
#pragma unroll
    for (int j = 0; j < VEC; ++j) {
        atomicAdd(ap + j, w * zp[j]);
    }
}

// z = (1-alpha)*agg + alpha*h, and clear agg for the next step. float4 wide.
__global__ void prop_update_kernel(float* __restrict__ agg,
                                   const float* __restrict__ h,
                                   float* __restrict__ z, int n4) {
    int i = blockIdx.x * blockDim.x + threadIdx.x;
    if (i >= n4) return;
    float4 a  = ((const float4*)agg)[i];
    float4 hv = ((const float4*)h)[i];
    float4 zv;
    zv.x = (1.0f - ALPHA) * a.x + ALPHA * hv.x;
    zv.y = (1.0f - ALPHA) * a.y + ALPHA * hv.y;
    zv.z = (1.0f - ALPHA) * a.z + ALPHA * hv.z;
    zv.w = (1.0f - ALPHA) * a.w + ALPHA * hv.w;
    ((float4*)z)[i] = zv;
    ((float4*)agg)[i] = make_float4(0.f, 0.f, 0.f, 0.f);
}

__global__ void zero4_kernel(float* __restrict__ p, int n4) {
    int i = blockIdx.x * blockDim.x + threadIdx.x;
    if (i < n4) ((float4*)p)[i] = make_float4(0.f, 0.f, 0.f, 0.f);
}

// ------------------------------- launcher ----------------------------------

static inline int cdiv(long long a, long long b) { return (int)((a + b - 1) / b); }

extern "C" void kernel_launch(void* const* d_in, const int* in_sizes, int n_in,
                              void* d_out, int out_size, void* d_ws, size_t ws_size,
                              hipStream_t stream) {
    (void)in_sizes; (void)n_in; (void)out_size; (void)ws_size;

    const float*     x   = (const float*)d_in[0];
    const long long* ei  = (const long long*)d_in[1];
    const float*     W3  = (const float*)d_in[2];
    const float*     b3  = (const float*)d_in[3];
    const float*     W4  = (const float*)d_in[4];
    const float*     b4  = (const float*)d_in[5];
    float*           out = (float*)d_out;

    const long long* src = ei;              // edge_index[0]
    const long long* dst = ei + N_EDGES;    // edge_index[1]

    // Workspace layout (all f32, 256B aligned). Total ~80.4 MB.
    char* ws = (char*)d_ws;
    size_t off = 0;
    auto alloc = [&](size_t bytes) {
        float* p = (float*)(ws + off);
        off += (bytes + 255) & ~(size_t)255;
        return p;
    };
    float* dinv = alloc((size_t)N_NODES * 4);
    float* norm = alloc((size_t)(N_EDGES + N_NODES) * 4);
    float* h1   = alloc((size_t)N_NODES * HID_CH * 4);   // reused as h2 later
    float* z1   = alloc((size_t)N_NODES * HID_CH * 4);
    float* agg  = alloc((size_t)N_NODES * HID_CH * 4);
    float* h2   = h1;   // h1 dead after phase-1 propagation

    const int EN = N_EDGES + N_NODES;

    // ---- GCN norm ----
    deg_init_kernel <<<cdiv(N_NODES, 256), 256, 0, stream>>>(dinv, N_NODES);
    deg_accum_kernel<<<cdiv(N_EDGES, 256), 256, 0, stream>>>(dst, dinv, N_EDGES);
    deg_fin_kernel  <<<cdiv(N_NODES, 256), 256, 0, stream>>>(dinv, N_NODES);
    norm_kernel     <<<cdiv(EN, 256), 256, 0, stream>>>(src, dst, dinv, norm,
                                                        N_EDGES, N_NODES);

    // ---- h1 = x @ W3 + b3   (WMMA bf16, 8 waves = 8 column tiles) ----
    gemm_wmma_kernel<IN_CH, HID_CH, false>
        <<<N_NODES / 16, 32 * (HID_CH / 16), 0, stream>>>(x, W3, b3, h1);

    // ---- APPNP phase 1 (128 channels) ----
    {
        const int n4 = N_NODES * HID_CH / 4;
        zero4_kernel<<<cdiv(n4, 256), 256, 0, stream>>>(agg, n4);
        const float* zcur = h1;
        const int scatterBlocks = cdiv((long long)EN * 32, 256);
        for (int k = 0; k < K_STEPS; ++k) {
            prop_scatter_kernel<HID_CH><<<scatterBlocks, 256, 0, stream>>>(
                zcur, src, dst, norm, agg, N_EDGES, N_NODES);
            prop_update_kernel<<<cdiv(n4, 256), 256, 0, stream>>>(agg, h1, z1, n4);
            zcur = z1;
        }
    }

    // ---- h2 = relu(z1) @ W4 + b4   (ReLU fused into A staging) ----
    gemm_wmma_kernel<HID_CH, OUT_CH, true>
        <<<N_NODES / 16, 32 * (OUT_CH / 16), 0, stream>>>(z1, W4, b4, h2);

    // ---- APPNP phase 2 (64 channels), z lives in d_out ----
    {
        const int n4 = N_NODES * OUT_CH / 4;
        zero4_kernel<<<cdiv(n4, 256), 256, 0, stream>>>(agg, n4);
        const float* zcur = h2;
        const int scatterBlocks = cdiv((long long)EN * 32, 256);
        for (int k = 0; k < K_STEPS; ++k) {
            prop_scatter_kernel<OUT_CH><<<scatterBlocks, 256, 0, stream>>>(
                zcur, src, dst, norm, agg, N_EDGES, N_NODES);
            prop_update_kernel<<<cdiv(n4, 256), 256, 0, stream>>>(agg, h2, out, n4);
            zcur = out;
        }
    }
}
